// MealShieldGNN_TDA_87806311399654
// MI455X (gfx1250) — compile-verified, matrix operand order checked by reading, not run on hardware
//
#include <hip/hip_runtime.h>
#include <hip/hip_bf16.h>
#include <cmath>

// ---------------- problem constants ----------------
#define NN      20000            // nodes
#define NE      320000           // edges (without self loops)
#define NEL     340000           // edges + self loops
#define NG      256              // graphs
#define FD      128              // hidden dim D
#define NH      4                // heads
#define HD      512              // H*D
#define NL      3                // layers
#define NT      6                // task heads

typedef __attribute__((ext_vector_type(16))) _Float16 v16h;
typedef __attribute__((ext_vector_type(8)))  _Float16 v8h;
typedef __attribute__((ext_vector_type(8)))  float    v8f;

// ---- float atomic max via sign-aware integer atomics (init to -inf) ----
__device__ __forceinline__ void atomicMaxF(float* addr, float val) {
  if (val >= 0.0f) {
    atomicMax((int*)addr, __float_as_int(val));
  } else {
    atomicMin((unsigned int*)addr, __float_as_uint(val));
  }
}

__device__ __forceinline__ float blockSum128(float v, float* red) {
  const unsigned t = threadIdx.x;
  red[t] = v;
  __syncthreads();
  for (unsigned s = 64; s > 0; s >>= 1) {
    if (t < s) red[t] += red[t + s];
    __syncthreads();
  }
  float r = red[0];
  __syncthreads();
  return r;
}

// ---------------- generic fill ----------------
__global__ void fill_kernel(float* __restrict__ p, int n, float v) {
  int gid = blockIdx.x * 256 + threadIdx.x;
  if (gid < n) p[gid] = v;
}

// ---------------- input projection: h = relu(x @ w_in + b_in) ----------------
__global__ void input_proj(const float* __restrict__ x, const float* __restrict__ w,
                           const float* __restrict__ b, float* __restrict__ h) {
  int gid = blockIdx.x * 256 + threadIdx.x;
  if (gid >= NN * FD) return;
  int n = gid >> 7, d = gid & 127;
  float s = b[d];
#pragma unroll
  for (int k = 0; k < 11; ++k) s += x[n * 11 + k] * w[k * FD + d];
  h[gid] = fmaxf(s, 0.0f);
}

// ---------------- once-per-layer f16 staging ----------------
// h [NN,128] f32 -> h16 [NN,128] f16 (WMMA A operand source, contiguous K)
__global__ void cvt_h_f16(const float* __restrict__ h, _Float16* __restrict__ h16) {
  int gid = blockIdx.x * 256 + threadIdx.x;
  if (gid < NN * FD) h16[gid] = (_Float16)h[gid];
}
// W [128,512] f32 -> Wt16 [512,128] f16 (transposed so B columns are contiguous in K)
__global__ void cvt_wT_f16(const float* __restrict__ W, _Float16* __restrict__ wt) {
  int gid = blockIdx.x * 256 + threadIdx.x;
  if (gid >= FD * HD) return;
  int j = gid >> 7, k = gid & 127;   // j: output col 0..511, k: K row 0..127
  wt[gid] = (_Float16)W[k * HD + j];
}

// ---------------- hh = h @ W   [20000,128] x [128,512], WMMA f16 -> f32 ----------------
// one wave computes a 16x16 tile (4x K=32 WMMA); block = 4 waves = 16 rows x 64 cols.
// Per K-chunk each lane needs two contiguous 8-half runs per operand -> 4x b128 loads.
__global__ void gat_gemm_wmma(const _Float16* __restrict__ h16,
                              const _Float16* __restrict__ wt,
                              float* __restrict__ hh) {
  const unsigned lane = threadIdx.x & 31u;
  const unsigned wave = threadIdx.x >> 5;        // 0..3
  const unsigned m0   = blockIdx.x * 16u;        // 1250 row tiles, exact
  const unsigned n0   = blockIdx.y * 64u + wave * 16u;
  const unsigned half = lane >> 4;               // K sub-block select per lane half
  const unsigned l16  = lane & 15u;

  const _Float16* arow = h16 + (size_t)(m0 + l16) * FD;   // A: M on lanes, K contiguous
  const _Float16* brow = wt  + (size_t)(n0 + l16) * FD;   // B: N on lanes, K contiguous

  v8f acc = {};
#pragma unroll
  for (unsigned kc = 0; kc < 128; kc += 32) {
    // 16-bit 16x32 fragment layout: lane half 0 holds K {0..7,16..23}, half 1 {8..15,24..31}
    v8h alo = *(const v8h*)(arow + kc + half * 8u);
    v8h ahi = *(const v8h*)(arow + kc + 16u + half * 8u);
    v8h blo = *(const v8h*)(brow + kc + half * 8u);
    v8h bhi = *(const v8h*)(brow + kc + 16u + half * 8u);
    v16h a, b;
#pragma unroll
    for (unsigned i = 0; i < 8; ++i) {
      a[i] = alo[i]; a[i + 8] = ahi[i];
      b[i] = blo[i]; b[i + 8] = bhi[i];
    }
    acc = __builtin_amdgcn_wmma_f32_16x16x32_f16(false, a, false, b, (short)0, acc,
                                                 false, false);
  }
  // 32-bit C/D 16x16: VGPR r -> M=r (lanes 0-15) / M=8+r (lanes 16-31), N=lane%16
  const unsigned moff = half * 8u;
#pragma unroll
  for (unsigned r = 0; r < 8; ++r)
    hh[(size_t)(m0 + moff + r) * HD + n0 + l16] = acc[r];
}

// ------- per-node attention logits: alpha_{s,d}[n,h] = <hh[n,h,:], a_{s,d}[h,:]> -------
__global__ void alpha_kernel(const float* __restrict__ hh, const float* __restrict__ a_s,
                             const float* __restrict__ a_d, float* __restrict__ alpha_s,
                             float* __restrict__ alpha_d) {
  __shared__ float red[128];
  const int n = blockIdx.x;
  const int t = threadIdx.x;
#pragma unroll
  for (int hh_i = 0; hh_i < NH; ++hh_i) {
    float v = hh[(size_t)n * HD + hh_i * FD + t];
    float s = blockSum128(v * a_s[hh_i * FD + t], red);
    if (t == 0) alpha_s[n * NH + hh_i] = s;
    float d = blockSum128(v * a_d[hh_i * FD + t], red);
    if (t == 0) alpha_d[n * NH + hh_i] = d;
  }
}

// ------- edge pass 1: e = leaky_relu(as[src]+ad[dst]); segment-max into emax -----------
__global__ void edge_logits(const int* __restrict__ ei, const float* __restrict__ as,
                            const float* __restrict__ ad, float* __restrict__ e_ws,
                            float* __restrict__ emax) {
  int gid = blockIdx.x * 256 + threadIdx.x;
  if (gid >= NEL * NH) return;
  int e = gid >> 2, hd = gid & 3;
  int s = (e < NE) ? ei[e] : (e - NE);
  int d = (e < NE) ? ei[NE + e] : (e - NE);
  float v = as[s * NH + hd] + ad[d * NH + hd];
  v = (v > 0.0f) ? v : 0.2f * v;     // leaky_relu, PyG slope
  e_ws[gid] = v;
  atomicMaxF(&emax[d * NH + hd], v);
}

// ------- edge pass 2: a = exp(e - emax[dst]); segment-sum into denom -------------------
__global__ void edge_exp(const int* __restrict__ ei, float* __restrict__ e_ws,
                         const float* __restrict__ emax, float* __restrict__ denom) {
  int gid = blockIdx.x * 256 + threadIdx.x;
  if (gid >= NEL * NH) return;
  int e = gid >> 2, hd = gid & 3;
  int d = (e < NE) ? ei[NE + e] : (e - NE);
  float a = __expf(e_ws[gid] - emax[d * NH + hd]);
  e_ws[gid] = a;                     // overwrite in place
  atomicAdd(&denom[d * NH + hd], a);
}

// ------- edge pass 3: agg[dst] += hh[src] * a/denom[dst] -------------------------------
// one thread per (edge, head, 4-wide d group): coalesced 4-float gather + 4 atomic adds;
// hh + agg (~82MB) stay L2-resident so this runs at L2/atomic-unit rate, not HBM rate.
__global__ void edge_agg_kernel(const int* __restrict__ ei, const float* __restrict__ e_ws,
                                const float* __restrict__ denom, const float* __restrict__ hh,
                                float* __restrict__ agg) {
  int gid = blockIdx.x * 256 + threadIdx.x;
  if (gid >= NEL * 128) return;
  int e   = gid >> 7;
  int rem = gid & 127;
  int hd  = rem >> 5;
  int d4  = (rem & 31) * 4;
  int s = (e < NE) ? ei[e] : (e - NE);
  int d = (e < NE) ? ei[NE + e] : (e - NE);
  float w = e_ws[e * NH + hd] / denom[d * NH + hd];
  const float* sp = hh + (size_t)s * HD + hd * FD + d4;
  float* dp = agg + (size_t)d * HD + hd * FD + d4;
  atomicAdd(dp + 0, sp[0] * w);
  atomicAdd(dp + 1, sp[1] * w);
  atomicAdd(dp + 2, sp[2] * w);
  atomicAdd(dp + 3, sp[3] * w);
}

// ------- node epilogue: head-mean + bias -> layernorm -> relu -> + residual (in place) -
__global__ void node_finalize(const float* __restrict__ agg, const float* __restrict__ bg,
                              const float* __restrict__ lnw, const float* __restrict__ lnb,
                              float* __restrict__ h) {
  __shared__ float red[128];
  const int n = blockIdx.x;
  const int t = threadIdx.x;
  const size_t base = (size_t)n * HD;
  float g = 0.25f * (agg[base + t] + agg[base + FD + t] + agg[base + 2 * FD + t] +
                     agg[base + 3 * FD + t]) + bg[t];
  float mu  = blockSum128(g, red) * (1.0f / 128.0f);
  float dv  = g - mu;
  float var = blockSum128(dv * dv, red) * (1.0f / 128.0f);
  float y   = dv * rsqrtf(var + 1e-5f) * lnw[t] + lnb[t];
  h[(size_t)n * FD + t] = fmaxf(y, 0.0f) + h[(size_t)n * FD + t];
}

// ---------------- per-graph pooling (sum / max / count via atomics) --------------------
__global__ void pool_accum(const float* __restrict__ h, const int* __restrict__ batch,
                           float* __restrict__ gsum, float* __restrict__ gmax,
                           float* __restrict__ gcnt) {
  const int n = blockIdx.x;
  const int t = threadIdx.x;
  const int b = batch[n];
  float v = h[(size_t)n * FD + t];
  atomicAdd(&gsum[b * FD + t], v);
  atomicMaxF(&gmax[b * FD + t], v);
  if (t == 0) atomicAdd(&gcnt[b], 1.0f);
}

__global__ void pool_finalize(const float* __restrict__ gsum, const float* __restrict__ gmax,
                              const float* __restrict__ gcnt, float* __restrict__ combined) {
  int gid = blockIdx.x * 256 + threadIdx.x;
  if (gid >= NG * FD) return;
  int g = gid >> 7, d = gid & 127;
  float c = gcnt[g];
  combined[g * 320 + d]      = gsum[gid] / fmaxf(c, 1.0f);
  combined[g * 320 + FD + d] = (c > 0.0f) ? gmax[gid] : 0.0f;
}

// ---------------- TDA branch: relu(relu(tda@w1+b1)@w2+b2) -> combined[:,256:320] -------
__global__ void tda_mlp(const float* __restrict__ tda, const float* __restrict__ w1,
                        const float* __restrict__ b1, const float* __restrict__ w2,
                        const float* __restrict__ b2, float* __restrict__ combined) {
  __shared__ float t1[64];
  const int g = blockIdx.x;
  const int j = threadIdx.x;   // 64 threads
  float s = b1[j];
#pragma unroll
  for (int k = 0; k < 30; ++k) s += tda[g * 30 + k] * w1[k * 64 + j];
  t1[j] = fmaxf(s, 0.0f);
  __syncthreads();
  float s2 = b2[j];
  for (int k = 0; k < 64; ++k) s2 += t1[k] * w2[k * 64 + j];
  combined[g * 320 + 256 + j] = fmaxf(s2, 0.0f);
}

// ---------------- shared trunk + 6 task heads, one block per graph ---------------------
__global__ void trunk_heads(const float* __restrict__ combined,
                            const float* __restrict__ w1, const float* __restrict__ b1,
                            const float* __restrict__ w2, const float* __restrict__ b2,
                            const float* __restrict__ wh1, const float* __restrict__ bh1,
                            const float* __restrict__ wh2, const float* __restrict__ bh2,
                            float* __restrict__ out) {
  __shared__ float comb[320];
  __shared__ float s1[256];
  __shared__ float s2[128];
  __shared__ float h1[64];
  __shared__ float red[64];
  const int g = blockIdx.x;
  const int t = threadIdx.x;   // 256 threads
  for (int i = t; i < 320; i += 256) comb[i] = combined[g * 320 + i];
  __syncthreads();
  float s = b1[t];
  for (int k = 0; k < 320; ++k) s += comb[k] * w1[k * 256 + t];
  s1[t] = fmaxf(s, 0.0f);
  __syncthreads();
  if (t < 128) {
    float v = b2[t];
    for (int k = 0; k < 256; ++k) v += s1[k] * w2[k * 128 + t];
    s2[t] = fmaxf(v, 0.0f);
  }
  __syncthreads();
  for (int tt = 0; tt < NT; ++tt) {
    if (t < 64) {
      float v = bh1[tt * 64 + t];
      for (int k = 0; k < 128; ++k) v += s2[k] * wh1[tt * 128 * 64 + k * 64 + t];
      h1[t] = fmaxf(v, 0.0f);
    }
    __syncthreads();
    if (t < 64) red[t] = h1[t] * wh2[tt * 64 + t];
    __syncthreads();
    for (int ss = 32; ss > 0; ss >>= 1) {
      if (t < ss) red[t] += red[t + ss];
      __syncthreads();
    }
    if (t == 0) out[tt * NG + g] = red[0] + bh2[tt];
    __syncthreads();
  }
}

// =======================================================================================
extern "C" void kernel_launch(void* const* d_in, const int* in_sizes, int n_in,
                              void* d_out, int out_size, void* d_ws, size_t ws_size,
                              hipStream_t stream) {
  (void)in_sizes; (void)n_in; (void)out_size; (void)ws_size;
  const float* x     = (const float*)d_in[0];
  const int*   ei    = (const int*)d_in[1];
  const int*   batch = (const int*)d_in[2];
  const float* tda   = (const float*)d_in[3];
  const float* w_in  = (const float*)d_in[4];
  const float* b_in  = (const float*)d_in[5];
  const float* w_gat = (const float*)d_in[6];
  const float* a_src = (const float*)d_in[7];
  const float* a_dst = (const float*)d_in[8];
  const float* b_gat = (const float*)d_in[9];
  const float* ln_w  = (const float*)d_in[10];
  const float* ln_b  = (const float*)d_in[11];
  const float* w_t1  = (const float*)d_in[12];
  const float* b_t1  = (const float*)d_in[13];
  const float* w_t2  = (const float*)d_in[14];
  const float* b_t2  = (const float*)d_in[15];
  const float* w_s1  = (const float*)d_in[16];
  const float* b_s1  = (const float*)d_in[17];
  const float* w_s2  = (const float*)d_in[18];
  const float* b_s2  = (const float*)d_in[19];
  const float* w_h1  = (const float*)d_in[20];
  const float* b_h1  = (const float*)d_in[21];
  const float* w_h2  = (const float*)d_in[22];
  const float* b_h2  = (const float*)d_in[23];
  float* out = (float*)d_out;

  // -------- workspace carve-up (float units) -------------------------------------------
  float* ws = (float*)d_ws;
  float* hh       = ws; ws += (size_t)NN * HD;        // 10,240,000
  float* agg      = ws; ws += (size_t)NN * HD;        // 10,240,000
  float* e_ws     = ws; ws += (size_t)NEL * NH;       //  1,360,000
  float* as       = ws; ws += (size_t)NN * NH;
  float* ad       = ws; ws += (size_t)NN * NH;
  float* emax     = ws; ws += (size_t)NN * NH;
  float* denom    = ws; ws += (size_t)NN * NH;
  float* gsum     = ws; ws += (size_t)NG * FD;
  float* gmax     = ws; ws += (size_t)NG * FD;
  float* gcnt     = ws; ws += NG;
  float* combined = ws; ws += (size_t)NG * 320;
  float* h        = ws; ws += (size_t)NN * FD;        //  2,560,000
  _Float16* h16   = (_Float16*)ws; ws += ((size_t)NN * FD) / 2;   // f16 staged A
  _Float16* wt16  = (_Float16*)ws; ws += ((size_t)FD * HD) / 2;   // f16 staged B^T

  const float NEG_INF = -INFINITY;

  input_proj<<<(NN * FD + 255) / 256, 256, 0, stream>>>(x, w_in, b_in, h);

  for (int l = 0; l < NL; ++l) {
    cvt_h_f16<<<(NN * FD + 255) / 256, 256, 0, stream>>>(h, h16);
    cvt_wT_f16<<<(FD * HD + 255) / 256, 256, 0, stream>>>(w_gat + (size_t)l * FD * HD,
                                                          wt16);
    gat_gemm_wmma<<<dim3(NN / 16, HD / 64), 128, 0, stream>>>(h16, wt16, hh);
    alpha_kernel<<<NN, 128, 0, stream>>>(hh, a_src + l * NH * FD, a_dst + l * NH * FD,
                                         as, ad);
    fill_kernel<<<(NN * NH + 255) / 256, 256, 0, stream>>>(emax, NN * NH, NEG_INF);
    fill_kernel<<<(NN * NH + 255) / 256, 256, 0, stream>>>(denom, NN * NH, 0.0f);
    fill_kernel<<<(NN * HD + 255) / 256, 256, 0, stream>>>(agg, NN * HD, 0.0f);
    edge_logits<<<(NEL * NH + 255) / 256, 256, 0, stream>>>(ei, as, ad, e_ws, emax);
    edge_exp<<<(NEL * NH + 255) / 256, 256, 0, stream>>>(ei, e_ws, emax, denom);
    edge_agg_kernel<<<(NEL * 128 + 255) / 256, 256, 0, stream>>>(ei, e_ws, denom, hh, agg);
    node_finalize<<<NN, 128, 0, stream>>>(agg, b_gat + l * FD, ln_w + l * FD,
                                          ln_b + l * FD, h);
  }

  fill_kernel<<<(NG * FD + 255) / 256, 256, 0, stream>>>(gsum, NG * FD, 0.0f);
  fill_kernel<<<(NG * FD + 255) / 256, 256, 0, stream>>>(gmax, NG * FD, NEG_INF);
  fill_kernel<<<1, 256, 0, stream>>>(gcnt, NG, 0.0f);
  pool_accum<<<NN, 128, 0, stream>>>(h, batch, gsum, gmax, gcnt);
  pool_finalize<<<(NG * FD + 255) / 256, 256, 0, stream>>>(gsum, gmax, gcnt, combined);
  tda_mlp<<<NG, 64, 0, stream>>>(tda, w_t1, b_t1, w_t2, b_t2, combined);
  trunk_heads<<<NG, 256, 0, stream>>>(combined, w_s1, b_s1, w_s2, b_s2, w_h1, b_h1,
                                      w_h2, b_h2, out);
}
